// EarthAttention3D_5729486373389
// MI455X (gfx1250) — matrix operand order repaired and checked by reading, not compiled, not run on hardware
//
#include <hip/hip_runtime.h>
#include <hip/hip_bf16.h>

typedef _Float16 half_t;
typedef __attribute__((ext_vector_type(16))) _Float16 v16h;
typedef __attribute__((ext_vector_type(8)))  _Float16 v8h;
typedef __attribute__((ext_vector_type(4)))  _Float16 v4h;
typedef __attribute__((ext_vector_type(8)))  float    v8f;

#define DIMC   192
#define HEADS  6
#define HD     32
#define TT     144
#define TPAD   160
#define NWIN   960
#define NSUB   60
#define ATT_SCALE 0.17677669529663687f   // 32^-0.5

#define USE_ASYNC_LDS 1

// workspace layout (element offsets in half_t units)
#define Q_OFF  0
#define K_OFF  (NWIN*HEADS*TT*HD)                  // 26,542,080
#define V_OFF  (2*(NWIN*HEADS*TT*HD))              // 53,084,160
#define O_OFF  (3*(NWIN*HEADS*TT*HD))              // 79,626,240
#define PB_OFF (O_OFF + NWIN*TT*DIMC)              // 106,168,320 (float region after)
#define PB_ELEMS ((long)NSUB*HEADS*TT*TT)          // 7,464,960 floats

// ---------------------------------------------------------------------------
// WMMA fragment loaders (contiguous-K order, same for A and B, wave32)
// A 16x32 f16: lane%16 = row, 16 contiguous K per lane half
// B 32x16 f16: lane%16 = column, 16 contiguous K per lane half
// ---------------------------------------------------------------------------
__device__ __forceinline__ v16h load_frag_a(const half_t* base, int stride, int row0, int k0) {
    int lane = threadIdx.x & 31;
    const half_t* p = base + (row0 + (lane & 15)) * stride + k0 + ((lane >> 4) << 4);
    return *(const v16h*)p;
}

// base is the B operand stored "column major": base[n*stride + k] == B(k, n)
__device__ __forceinline__ v16h load_frag_b(const half_t* base, int stride, int n0, int k0) {
    int lane = threadIdx.x & 31;
    const half_t* p = base + (n0 + (lane & 15)) * stride + k0 + ((lane >> 4) << 4);
    return *(const v16h*)p;
}

__device__ __forceinline__ v8f wmma_f16(v16h a, v16h b, v8f c) {
    return __builtin_amdgcn_wmma_f32_16x16x32_f16(false, a, false, b, (short)0, c, false, false);
}

// ---------------------------------------------------------------------------
// Precompute earth-position bias table pbt[sub][h][t][s] (29.9 MB, L2-resident)
// so the hot attention kernel reads it with coalesced streaming loads.
// ---------------------------------------------------------------------------
__global__ __launch_bounds__(256)
void ea_pb_kernel(const float* __restrict__ btab, float* __restrict__ pbt)
{
    long e = (long)blockIdx.x * blockDim.x + threadIdx.x;
    if (e >= PB_ELEMS) return;
    int s = (int)(e % TT); long r = e / TT;
    int t = (int)(r % TT); r /= TT;
    int h = (int)(r % HEADS);
    int sub = (int)(r / HEADS);
    int zt = t / 72, ht = (t % 72) / 12, wt = t % 12;
    int zs = s / 72, hs = (s % 72) / 12, wsd = s % 12;
    int idx = (zt + zs * 2) * 1584 + (ht - hs + 5) * 144 + (wt + wsd * 12);
    pbt[e] = btab[((long)idx * NSUB + sub) * HEADS + h];
}

// ---------------------------------------------------------------------------
// Kernel A: qkv = f16(x @ W_qkv + b_qkv), scattered into Q/K/V workspaces
// grid (576/64, 138240/64), 256 threads (8 waves, 2 output tiles each)
// ---------------------------------------------------------------------------
__global__ __launch_bounds__(256)
void ea_qkv_kernel(const float* __restrict__ x, const float* __restrict__ Wq,
                   const float* __restrict__ bq, half_t* __restrict__ ws)
{
    __shared__ __align__(32) half_t As[64 * 192];   // x tile, f16
    __shared__ __align__(32) half_t Bt[64 * 192];   // W tile, transposed f16
    const int tid = threadIdx.x;
    const int n0  = blockIdx.x * 64;
    const int m0  = blockIdx.y * 64;

    const float4* x4 = (const float4*)x;
#pragma unroll
    for (int i = 0; i < 12; ++i) {
        int idx = i * 256 + tid;                 // 0..3071
        int row = idx / 48, c4 = idx % 48;       // 48 float4 per row
        float4 v = x4[(long)(m0 + row) * 48 + c4];
        v4h h; h[0] = (half_t)v.x; h[1] = (half_t)v.y; h[2] = (half_t)v.z; h[3] = (half_t)v.w;
        *(v4h*)(&As[row * 192 + c4 * 4]) = h;
    }
#pragma unroll
    for (int i = 0; i < 12; ++i) {
        int idx = i * 256 + tid;
        int k = idx / 16, c4 = idx % 16;
        const float4* wrow = (const float4*)(Wq + (long)k * 576 + n0);
        float4 v = wrow[c4];
        Bt[(c4 * 4 + 0) * 192 + k] = (half_t)v.x;
        Bt[(c4 * 4 + 1) * 192 + k] = (half_t)v.y;
        Bt[(c4 * 4 + 2) * 192 + k] = (half_t)v.z;
        Bt[(c4 * 4 + 3) * 192 + k] = (half_t)v.w;
    }
    __syncthreads();

    const int wv = tid >> 5;
    const int tr = (wv >> 1) * 16;
    const int tc = (wv & 1) * 32;
    v8f acc0 = {}, acc1 = {};
#pragma unroll
    for (int kk = 0; kk < 6; ++kk) {
        v16h a  = load_frag_a(As, 192, tr, kk * 32);
        v16h b0 = load_frag_b(Bt, 192, tc,      kk * 32);
        acc0 = wmma_f16(a, b0, acc0);
        v16h b1 = load_frag_b(Bt, 192, tc + 16, kk * 32);
        acc1 = wmma_f16(a, b1, acc1);
    }

    const int lane = tid & 31;
    const int ncol = lane & 15;
    const int hi8  = (lane >> 4) * 8;
    half_t* qws = ws + Q_OFF;
    half_t* kws = ws + K_OFF;
    half_t* vws = ws + V_OFF;
#pragma unroll
    for (int c = 0; c < 2; ++c) {
        v8f acc = c ? acc1 : acc0;
        int cg = n0 + tc + c * 16 + ncol;
        float bias = bq[cg];
        int which = cg / 192;            // 0=q 1=k 2=v (uniform per 16-col tile)
        int hh = (cg % 192) / HD;
        int dd = cg % HD;
#pragma unroll
        for (int g = 0; g < 8; ++g) {
            int m = m0 + tr + hi8 + g;
            int b = m / TT, t = m % TT;
            half_t hv = (half_t)(acc[g] + bias);
            long base = ((long)b * HEADS + hh) * TT + t;
            if (which == 0)      qws[base * HD + dd] = hv;
            else if (which == 1) kws[base * HD + dd] = hv;
            else                 vws[base * HD + dd] = hv;   // row-major like K
        }
    }
}

// ---------------------------------------------------------------------------
// Kernel B: fused attention per (window, head):
//   S = scale*QK^T + pbt + mask; P = softmax(S); O = P @ V
// grid 5760, 288 threads (9 waves, one 16-row strip each)
// ---------------------------------------------------------------------------
__global__ __launch_bounds__(288)
void ea_attn_kernel(const float* __restrict__ mask, const float* __restrict__ pbt,
                    half_t* __restrict__ ws)
{
    __shared__ __align__(32) half_t Qs[TT * HD];      // 144x32
    __shared__ __align__(32) half_t Ks[TT * HD];      // 144x32
    __shared__ __align__(32) half_t Vs[TT * HD];      // 144x32 (row major)
    __shared__ __align__(32) half_t Vt[HD * TPAD];    // 32x160 (V^T, zero-padded)
    __shared__ __align__(32) half_t Ps[TT * TPAD];    // 144x160 P, zero-padded cols

    const int tid = threadIdx.x;
    const int bh  = blockIdx.x;
    const int b   = bh / HEADS, h = bh % HEADS;

    const half_t* qg = ws + Q_OFF + (long)bh * TT * HD;
    const half_t* kg = ws + K_OFF + (long)bh * TT * HD;
    const half_t* vg = ws + V_OFF + (long)bh * TT * HD;

#if USE_ASYNC_LDS
    // ASYNCcnt-tracked memory->LDS DMA (CDNA5 §10 async ops); 16B per lane.
    {
        uint32_t lq = (uint32_t)(uintptr_t)Qs;
        uint32_t lk = (uint32_t)(uintptr_t)Ks;
        uint32_t lv = (uint32_t)(uintptr_t)Vs;
        for (int i = tid; i < TT * HD / 8; i += 288) {   // 576 chunks of 8 halfs
            asm volatile("global_load_async_to_lds_b128 %0, %1, off"
                         :: "v"(lq + i * 16), "v"(qg + i * 8) : "memory");
            asm volatile("global_load_async_to_lds_b128 %0, %1, off"
                         :: "v"(lk + i * 16), "v"(kg + i * 8) : "memory");
            asm volatile("global_load_async_to_lds_b128 %0, %1, off"
                         :: "v"(lv + i * 16), "v"(vg + i * 8) : "memory");
        }
        asm volatile("s_wait_asynccnt 0" ::: "memory");
    }
#else
    {
        const uint4* q4 = (const uint4*)qg;
        const uint4* k4 = (const uint4*)kg;
        const uint4* v4 = (const uint4*)vg;
        uint4* qs4 = (uint4*)Qs; uint4* ks4 = (uint4*)Ks; uint4* vs4 = (uint4*)Vs;
        for (int i = tid; i < TT * HD / 8; i += 288) {
            qs4[i] = q4[i]; ks4[i] = k4[i]; vs4[i] = v4[i];
        }
    }
#endif
    __syncthreads();

    // in-LDS transpose V -> Vt[32][160] with zero padding for t=144..159
    for (int i = tid; i < TT * HD; i += 288) Vt[(i & 31) * TPAD + (i >> 5)] = Vs[i];
    for (int i = tid; i < HD * (TPAD - TT); i += 288)
        Vt[(i >> 4) * TPAD + TT + (i & 15)] = (half_t)0.f;

    const int wv   = tid >> 5;
    const int tb   = wv * 16;
    const int lane = tid & 31;
    const int ncol = lane & 15;
    const int hi8  = (lane >> 4) * 8;

    // S strip = Q[tb:tb+16] @ K^T  (K=32 in one WMMA per column tile)
    v16h aq = load_frag_a(Qs, HD, tb, 0);
    v8f S[9] = {};
#pragma unroll
    for (int j = 0; j < 9; ++j) {
        v16h bk = load_frag_b(Ks, HD, j * 16, 0);   // Ks[n*32+k] == (K^T)(k,n)
        S[j] = wmma_f16(aq, bk, S[j]);
    }

    // + precomputed earth bias + mask (both coalesced streaming reads)
    const int sub = (b / 240) * 15 + (b % 15);      // wz*NWW + ww
    const float* pbw = pbt + ((long)(sub * HEADS + h)) * TT * TT;
    const float* mw  = mask + (long)b * TT * TT;
#pragma unroll
    for (int g = 0; g < 8; ++g) {
        int t = tb + hi8 + g;
        const float* pr = pbw + (long)t * TT;
        const float* mr = mw  + (long)t * TT;
#pragma unroll
        for (int j = 0; j < 9; ++j) {
            int s = j * 16 + ncol;
            S[j][g] = S[j][g] * ATT_SCALE + pr[s] + mr[s];
        }
    }

    // softmax: row = tb+hi8+g lives across 16 lanes of one half-wave;
    // xor masks 1..8 reduce each half independently (wave32)
    float rinv[8];
#pragma unroll
    for (int g = 0; g < 8; ++g) {
        float m = -3.4e38f;
#pragma unroll
        for (int j = 0; j < 9; ++j) m = fmaxf(m, S[j][g]);
#pragma unroll
        for (int off = 1; off < 16; off <<= 1) m = fmaxf(m, __shfl_xor(m, off, 32));
        float sum = 0.f;
#pragma unroll
        for (int j = 0; j < 9; ++j) { float e = __expf(S[j][g] - m); S[j][g] = e; sum += e; }
#pragma unroll
        for (int off = 1; off < 16; off <<= 1) sum += __shfl_xor(sum, off, 32);
        rinv[g] = 1.f / sum;
    }

    // write own P strip (padded to 160 cols) — only this wave reads it back
#pragma unroll
    for (int g = 0; g < 8; ++g) {
        int row = tb + hi8 + g;
#pragma unroll
        for (int j = 0; j < 9; ++j)
            Ps[row * TPAD + j * 16 + ncol] = (half_t)(S[j][g] * rinv[g]);
        Ps[row * TPAD + 144 + ncol] = (half_t)0.f;
    }

    __syncthreads();   // Vt (written by all threads) must be complete before PV

    // O strip = P @ V  (K padded to 160 -> 5 steps; 2 column tiles of 16)
    v8f o0 = {}, o1 = {};
#pragma unroll
    for (int kk = 0; kk < 5; ++kk) {
        v16h ap = load_frag_a(Ps, TPAD, tb, kk * 32);
        v16h b0 = load_frag_b(Vt, TPAD, 0,  kk * 32);   // Vt[n*160+k] == V(k,n)
        o0 = wmma_f16(ap, b0, o0);
        v16h b1 = load_frag_b(Vt, TPAD, 16, kk * 32);
        o1 = wmma_f16(ap, b1, o1);
    }
    half_t* ows = ws + O_OFF;   // f16 [B_*T][192], head-interleaved layout
#pragma unroll
    for (int g = 0; g < 8; ++g) {
        long m = (long)b * TT + tb + hi8 + g;
        ows[m * DIMC + h * HD + ncol]      = (half_t)o0[g];
        ows[m * DIMC + h * HD + 16 + ncol] = (half_t)o1[g];
    }
}

// ---------------------------------------------------------------------------
// Kernel C: out = f32(o_ws(f16) @ W_out + b_out)
// grid (192/64, 138240/64), 256 threads
// ---------------------------------------------------------------------------
__global__ __launch_bounds__(256)
void ea_outproj_kernel(const float* __restrict__ Wo, const float* __restrict__ bo,
                       const half_t* __restrict__ ws, float* __restrict__ out)
{
    __shared__ __align__(32) half_t As[64 * 192];
    __shared__ __align__(32) half_t Bt[64 * 192];
    const int tid = threadIdx.x;
    const int n0  = blockIdx.x * 64;
    const int m0  = blockIdx.y * 64;
    const half_t* ows = ws + O_OFF;

    const uint4* a4 = (const uint4*)(ows + (long)m0 * 192);
    uint4* as4 = (uint4*)As;
#pragma unroll
    for (int i = 0; i < 6; ++i) as4[i * 256 + tid] = a4[i * 256 + tid];
#pragma unroll
    for (int i = 0; i < 12; ++i) {
        int idx = i * 256 + tid;
        int k = idx / 16, c4 = idx % 16;
        const float4* wrow = (const float4*)(Wo + (long)k * 192 + n0);
        float4 v = wrow[c4];
        Bt[(c4 * 4 + 0) * 192 + k] = (half_t)v.x;
        Bt[(c4 * 4 + 1) * 192 + k] = (half_t)v.y;
        Bt[(c4 * 4 + 2) * 192 + k] = (half_t)v.z;
        Bt[(c4 * 4 + 3) * 192 + k] = (half_t)v.w;
    }
    __syncthreads();

    const int wv = tid >> 5;
    const int tr = (wv >> 1) * 16;
    const int tc = (wv & 1) * 32;
    v8f acc0 = {}, acc1 = {};
#pragma unroll
    for (int kk = 0; kk < 6; ++kk) {
        v16h a  = load_frag_a(As, 192, tr, kk * 32);
        v16h b0 = load_frag_b(Bt, 192, tc,      kk * 32);
        acc0 = wmma_f16(a, b0, acc0);
        v16h b1 = load_frag_b(Bt, 192, tc + 16, kk * 32);
        acc1 = wmma_f16(a, b1, acc1);
    }

    const int lane = tid & 31;
    const int ncol = lane & 15;
    const int hi8  = (lane >> 4) * 8;
#pragma unroll
    for (int c = 0; c < 2; ++c) {
        v8f acc = c ? acc1 : acc0;
        int cg = n0 + tc + c * 16 + ncol;
        float bias = bo[cg];
#pragma unroll
        for (int g = 0; g < 8; ++g) {
            long m = (long)m0 + tr + hi8 + g;
            out[m * DIMC + cg] = acc[g] + bias;
        }
    }
}

// ---------------------------------------------------------------------------
extern "C" void kernel_launch(void* const* d_in, const int* in_sizes, int n_in,
                              void* d_out, int out_size, void* d_ws, size_t ws_size,
                              hipStream_t stream) {
    const float* x    = (const float*)d_in[0];
    const float* mask = (const float*)d_in[1];
    const float* Wq   = (const float*)d_in[2];
    const float* bq   = (const float*)d_in[3];
    const float* Wo   = (const float*)d_in[4];
    const float* bo   = (const float*)d_in[5];
    const float* btab = (const float*)d_in[6];
    half_t* ws  = (half_t*)d_ws;
    float*  out = (float*)d_out;
    float*  pbt = (float*)(ws + PB_OFF);

    // expand bias table -> pbt[sub][h][t][s] (L2-resident, read coalesced later)
    ea_pb_kernel<<<(int)((PB_ELEMS + 255) / 256), 256, 0, stream>>>(btab, pbt);

    // QKV projection: M=138240, N=576, K=192
    ea_qkv_kernel<<<dim3(576 / 64, 138240 / 64), 256, 0, stream>>>(x, Wq, bq, ws);

    // fused attention per (window, head)
    ea_attn_kernel<<<NWIN * HEADS, 288, 0, stream>>>(mask, pbt, ws);

    // output projection: M=138240, N=192, K=192
    ea_outproj_kernel<<<dim3(192 / 64, 138240 / 64), 256, 0, stream>>>(Wo, bo, ws, out);
}
